// ASTEnc_5566277616398
// MI455X (gfx1250) — compile-verified, matrix operand order checked by reading, not run on hardware
//
#include <hip/hip_runtime.h>

#define N_NODES 262144
#define N_EDGES 524288
#define EMB     256
#define LN_EPS  1e-5f

typedef __attribute__((ext_vector_type(16))) __bf16 v16bf;
typedef __attribute__((ext_vector_type(8)))  float  v8f;

union Frag {
    v16bf v;
    uint4 q[2];
    unsigned short s[16];
};

__device__ __forceinline__ unsigned short f2bf(float f) {
    union { float f; unsigned int u; } x; x.f = f;
    unsigned int u = x.u;
    unsigned int r = u + 0x7fffu + ((u >> 16) & 1u);   // round-to-nearest-even
    return (unsigned short)(r >> 16);
}

// ---------------------------------------------------------------------------
// Kernel 1: embedding gather + scale + pos-emb + LayerNorm.  1 wave per node.
// ---------------------------------------------------------------------------
__global__ void __launch_bounds__(256)
embed_ln_kernel(const int* __restrict__ ids, const int* __restrict__ pos,
                const float* __restrict__ ntab, const float* __restrict__ ptab,
                const float* __restrict__ g, const float* __restrict__ b,
                float* __restrict__ h) {
    const int node = blockIdx.x * 8 + (threadIdx.x >> 5);
    const int lane = threadIdx.x & 31;
    const int id = ids[node];
    const int p  = pos[node];
    const float* nr = ntab + (size_t)id * EMB + lane * 8;
    const float* pr = ptab + (size_t)p  * EMB + lane * 8;
    float v[8], s = 0.f, ss = 0.f;
#pragma unroll
    for (int j = 0; j < 8; ++j) {
        v[j] = nr[j] * 16.0f + pr[j];          // sqrt(EMB) = 16
        s += v[j]; ss += v[j] * v[j];
    }
#pragma unroll
    for (int m = 16; m; m >>= 1) {
        s  += __shfl_xor(s,  m, 32);
        ss += __shfl_xor(ss, m, 32);
    }
    const float mean = s * (1.f / EMB);
    const float var  = ss * (1.f / EMB) - mean * mean;
    const float inv  = rsqrtf(var + LN_EPS);
    float* o = h + (size_t)node * EMB + lane * 8;
#pragma unroll
    for (int j = 0; j < 8; ++j) {
        const int c = lane * 8 + j;
        o[j] = (v[j] - mean) * inv * g[c] + b[c];
    }
}

// ---------------------------------------------------------------------------
// Kernel 2: zero the aggregation buffer (float4 stores).
// ---------------------------------------------------------------------------
__global__ void zero4_kernel(float4* __restrict__ p, int n4) {
    const int i = blockIdx.x * blockDim.x + threadIdx.x;
    if (i < n4) p[i] = make_float4(0.f, 0.f, 0.f, 0.f);
}

// ---------------------------------------------------------------------------
// Kernel 3: edge scatter  agg[dst] += h[src].  One 64-thread block per edge,
// each thread moves a float4 and issues 4 native global_atomic_add_f32.
// ---------------------------------------------------------------------------
__global__ void __launch_bounds__(64)
sage_scatter_kernel(const float* __restrict__ h, const int* __restrict__ src,
                    const int* __restrict__ dst, float* __restrict__ agg) {
    const int e = blockIdx.x;
    const int t = threadIdx.x;          // 0..63 -> 4 floats each
    const int s = src[e];
    const int d = dst[e];
    const float4 v = *reinterpret_cast<const float4*>(h + (size_t)s * EMB + t * 4);
    float* a = agg + (size_t)d * EMB + t * 4;
    unsafeAtomicAdd(a + 0, v.x);
    unsafeAtomicAdd(a + 1, v.y);
    unsafeAtomicAdd(a + 2, v.z);
    unsafeAtomicAdd(a + 3, v.w);
}

// ---------------------------------------------------------------------------
// Kernel 4: pack [Wl^T ; Wr^T] (K=512 x N=256) into bf16 WMMA B-fragments.
// One wave per (kt,nt) tile; lane's 16 elements land as one 32B record.
//   lanes 0-15  : K = {0..7, 16..23} (+32*kt),   N = lane
//   lanes 16-31 : K = {8..15, 24..31} (+32*kt),  N = lane-16
// ---------------------------------------------------------------------------
__global__ void __launch_bounds__(32)
prep_w_kernel(const float* __restrict__ Wl, const float* __restrict__ Wr,
              uint4* __restrict__ out) {
    const int nt = blockIdx.x & 15;
    const int kt = blockIdx.x >> 4;
    const int lane = threadIdx.x;
    const int n  = nt * 16 + (lane & 15);
    const int kb = kt * 32 + ((lane >> 4) * 8);
    Frag f;
#pragma unroll
    for (int j = 0; j < 16; ++j) {
        const int k = kb + ((j < 8) ? j : (j + 8));
        // B[k][n] = Wl[n][k] (k<256) else Wr[n][k-256]   (out = A @ W^T)
        const float v = (k < EMB) ? Wl[n * EMB + k] : Wr[n * EMB + (k - EMB)];
        f.s[j] = f2bf(v);
    }
    const int fid = (kt * 16 + nt) * 32 + lane;
    out[fid * 2 + 0] = f.q[0];
    out[fid * 2 + 1] = f.q[1];
}

// ---------------------------------------------------------------------------
// GEMM fragment loaders.
// ---------------------------------------------------------------------------
__device__ __forceinline__ void load_a_frag(const unsigned short* abuf,
                                            int m, int khoff, int kt, Frag& a) {
    const int run0 = kt * 32 + khoff;
    a.q[0] = *reinterpret_cast<const uint4*>(&abuf[m * 520 + run0]);
    a.q[1] = *reinterpret_cast<const uint4*>(&abuf[m * 520 + run0 + 16]);
}

__device__ __forceinline__ void load_b_frags(const uint4* __restrict__ wfrag,
                                             int kt, int wave, int lane, Frag* bf) {
#pragma unroll
    for (int j = 0; j < 4; ++j) {
        const uint4* wp = wfrag + ((size_t)(kt * 16 + wave * 4 + j) * 32 + lane) * 2;
        bf[j].q[0] = wp[0];
        bf[j].q[1] = wp[1];
    }
}

// ---------------------------------------------------------------------------
// Kernel 5: fused concat-GEMM (K=512) + bias + ReLU + residual + LayerNorm.
// Block = 128 threads (4 waves) owns 16 rows x 256 cols.  Per wave: 4 N-tiles,
// 16 K-steps => 64 v_wmma_f32_16x16x32_bf16 per wave.  Software-pipelined:
// next K-step's A (LDS) and B (L2) fragments prefetch under current WMMAs.
// ---------------------------------------------------------------------------
__global__ void __launch_bounds__(128)
sage_gemm_ln_kernel(const float* __restrict__ agg, float* __restrict__ h,
                    const uint4* __restrict__ wfrag,
                    const float* __restrict__ bl, const float* __restrict__ g,
                    const float* __restrict__ b) {
    __shared__ __align__(16) unsigned short abuf[16 * 520];  // bf16 [16][512] pad 8
    __shared__ float ybuf[16 * 260];                         // f32  [16][256] pad 4
    const int tid = threadIdx.x;
    const int rowbase = blockIdx.x * 16;

    // Stage A = [agg | h] as bf16 into LDS.
    for (int idx = tid; idx < 16 * 512; idx += 128) {
        const int row = idx >> 9, col = idx & 511;
        const float v = (col < EMB)
            ? agg[(size_t)(rowbase + row) * EMB + col]
            : h  [(size_t)(rowbase + row) * EMB + (col - EMB)];
        abuf[row * 520 + col] = f2bf(v);
    }
    __syncthreads();

    const int wave = tid >> 5, lane = tid & 31;
    const int m     = lane & 15;
    const int khoff = (lane >> 4) * 8;

    v8f acc[4] = {};
    Frag a, an;
    Frag bf[4], bn[4];
    load_a_frag(abuf, m, khoff, 0, a);
    load_b_frags(wfrag, 0, wave, lane, bf);

#pragma unroll
    for (int kt = 0; kt < 16; ++kt) {
        if (kt < 15) {                          // prefetch next K-step
            load_a_frag(abuf, m, khoff, kt + 1, an);
            load_b_frags(wfrag, kt + 1, wave, lane, bn);
        }
#pragma unroll
        for (int j = 0; j < 4; ++j) {
            acc[j] = __builtin_amdgcn_wmma_f32_16x16x32_bf16(
                false, a.v, false, bf[j].v, (short)0, acc[j], false, false);
        }
        if (kt < 15) {
            a = an;
#pragma unroll
            for (int j = 0; j < 4; ++j) bf[j] = bn[j];
        }
    }

    // y = relu(gemm + bias) + h  -> LDS  (C/D layout: lane<16 M=r, lane>=16 M=r+8)
    const int mb = (lane >> 4) * 8;
#pragma unroll
    for (int j = 0; j < 4; ++j) {
        const int n = (wave * 4 + j) * 16 + (lane & 15);
        const float bias = bl[n];
#pragma unroll
        for (int r = 0; r < 8; ++r) {
            const int mm = r + mb;
            float y = acc[j][r] + bias;
            y = fmaxf(y, 0.0f) + h[(size_t)(rowbase + mm) * EMB + n];
            ybuf[mm * 260 + n] = y;
        }
    }
    __syncthreads();

    // LayerNorm: 8 threads per row, shuffle-reduce within aligned groups of 8.
    const int row = tid >> 3;
    const int cb  = (tid & 7) * 32;
    float s = 0.f, ss = 0.f;
    for (int j = 0; j < 32; ++j) {
        const float y = ybuf[row * 260 + cb + j];
        s += y; ss += y * y;
    }
#pragma unroll
    for (int msk = 4; msk; msk >>= 1) {
        s  += __shfl_xor(s,  msk, 8);
        ss += __shfl_xor(ss, msk, 8);
    }
    const float mean = s * (1.f / EMB);
    const float var  = ss * (1.f / EMB) - mean * mean;
    const float inv  = rsqrtf(var + LN_EPS);
    for (int j = 0; j < 32; ++j) {
        const int n = cb + j;
        const float y = ybuf[row * 260 + n];
        h[(size_t)(rowbase + row) * EMB + n] = (y - mean) * inv * g[n] + b[n];
    }
}

// ---------------------------------------------------------------------------
extern "C" void kernel_launch(void* const* d_in, const int* in_sizes, int n_in,
                              void* d_out, int out_size, void* d_ws, size_t ws_size,
                              hipStream_t stream) {
    (void)in_sizes; (void)n_in; (void)out_size; (void)ws_size;
    const int*   node_emb = (const int*)  d_in[0];
    const int*   pos      = (const int*)  d_in[1];
    const int*   edge     = (const int*)  d_in[2];
    const float* node_tab = (const float*)d_in[3];
    const float* pos_tab  = (const float*)d_in[4];
    const float* g_emb    = (const float*)d_in[5];
    const float* b_emb    = (const float*)d_in[6];
    const float* Wl0 = (const float*)d_in[7];
    const float* bl0 = (const float*)d_in[8];
    const float* Wr0 = (const float*)d_in[9];
    const float* g0  = (const float*)d_in[10];
    const float* b0  = (const float*)d_in[11];
    const float* Wl1 = (const float*)d_in[12];
    const float* bl1 = (const float*)d_in[13];
    const float* Wr1 = (const float*)d_in[14];
    const float* g1  = (const float*)d_in[15];
    const float* b1  = (const float*)d_in[16];

    float* h = (float*)d_out;                       // h lives in d_out throughout

    char*  ws  = (char*)d_ws;
    float* agg = (float*)ws;                                       // 256 MB
    uint4* wf0 = (uint4*)(ws + (size_t)N_NODES * EMB * sizeof(float));
    uint4* wf1 = wf0 + (size_t)16 * 16 * 32 * 2;                   // 256 KB each

    const int* src = edge;
    const int* dst = edge + N_EDGES;

    embed_ln_kernel<<<N_NODES / 8, 256, 0, stream>>>(
        node_emb, pos, node_tab, pos_tab, g_emb, b_emb, h);
    prep_w_kernel<<<256, 32, 0, stream>>>(Wl0, Wr0, wf0);
    prep_w_kernel<<<256, 32, 0, stream>>>(Wl1, Wr1, wf1);

    const float* BL[2] = {bl0, bl1};
    const float* G [2] = {g0,  g1};
    const float* B_[2] = {b0,  b1};
    uint4*       WF[2] = {wf0, wf1};

    for (int L = 0; L < 2; ++L) {
        zero4_kernel<<<(N_NODES * EMB / 4) / 256, 256, 0, stream>>>(
            (float4*)agg, N_NODES * EMB / 4);
        sage_scatter_kernel<<<N_EDGES, 64, 0, stream>>>(h, src, dst, agg);
        sage_gemm_ln_kernel<<<N_NODES / 16, 128, 0, stream>>>(
            agg, h, WF[L], BL[L], G[L], B_[L]);
    }
}
